// RLAAttention_27152783245538
// MI455X (gfx1250) — matrix operand order repaired and checked by reading, not compiled, not run on hardware
//
#include <hip/hip_runtime.h>
#include <hip/hip_bf16.h>
#include <math.h>

// ---------------------------------------------------------------------------
// Problem constants
// ---------------------------------------------------------------------------
#define BATCH     2
#define SEQ       2048
#define DMODEL    4096
#define NHEADS    32
#define NKVHEADS  8
#define HEADDIM   128
#define QDIM      (NHEADS * HEADDIM)       // 4096
#define KVDIM     (NKVHEADS * HEADDIM)     // 1024
#define QKVN      (QDIM + 2 * KVDIM)       // 6144
#define MROWS     (BATCH * SEQ)            // 4096
#define LOG2_THETA 18.931568569324174f     // log2(500000)
#define INV_SQRT_HD 0.08838834764831845f   // 1/sqrt(128)

typedef __attribute__((ext_vector_type(16))) __bf16          v16bf;
typedef __attribute__((ext_vector_type(8)))  float           v8f;
typedef __attribute__((ext_vector_type(8)))  unsigned short  u16x8;

union BF16Frag { v16bf v; unsigned short u[16]; };

__device__ __forceinline__ unsigned short f2bf(float f) {
    unsigned int u = __builtin_bit_cast(unsigned int, f);
    unsigned int r = u + 0x7FFFu + ((u >> 16) & 1u);   // round-to-nearest-even
    return (unsigned short)(r >> 16);
}

__device__ __forceinline__ v8f wmma_bf16(v16bf a, v16bf b, v8f c) {
    return __builtin_amdgcn_wmma_f32_16x16x32_bf16(false, a, false, b,
                                                   (short)0, c, false, false);
}

__device__ __forceinline__ void wait_async0() {
#if __has_builtin(__builtin_amdgcn_s_wait_asynccnt)
    __builtin_amdgcn_s_wait_asynccnt(0);
#else
    asm volatile("s_wait_asynccnt 0" ::: "memory");
#endif
}

// Async global->LDS copy of 128 bytes (8 x B128). IOFFSET is added to BOTH
// the LDS and global addresses, so one base pair covers the whole stripe.
__device__ __forceinline__ void async_copy_128B(unsigned lds_addr,
                                                unsigned long long gaddr) {
    asm volatile(
        "global_load_async_to_lds_b128 %0, %1, off\n\t"
        "global_load_async_to_lds_b128 %0, %1, off offset:16\n\t"
        "global_load_async_to_lds_b128 %0, %1, off offset:32\n\t"
        "global_load_async_to_lds_b128 %0, %1, off offset:48\n\t"
        "global_load_async_to_lds_b128 %0, %1, off offset:64\n\t"
        "global_load_async_to_lds_b128 %0, %1, off offset:80\n\t"
        "global_load_async_to_lds_b128 %0, %1, off offset:96\n\t"
        "global_load_async_to_lds_b128 %0, %1, off offset:112"
        :: "v"(lds_addr), "v"(gaddr) : "memory");
}

__device__ __forceinline__ void async_copy_32B(unsigned lds_addr,
                                               unsigned long long gaddr) {
    asm volatile(
        "global_load_async_to_lds_b128 %0, %1, off\n\t"
        "global_load_async_to_lds_b128 %0, %1, off offset:16"
        :: "v"(lds_addr), "v"(gaddr) : "memory");
}

// A-fragment (16x32 bf16): lane holds row M=lane%16;
// elems 0..15 = K (lane/16)*16+0..15.
__device__ __forceinline__ v16bf load_a_frag(const unsigned short* rowbase, int koff) {
    BF16Frag f;
    u16x8 lo = *(const u16x8*)(rowbase + 2 * koff);
    u16x8 hi = *(const u16x8*)(rowbase + 2 * koff + 8);
#pragma unroll
    for (int e = 0; e < 8; ++e) { f.u[e] = lo[e]; f.u[8 + e] = hi[e]; }
    return f.v;
}

// B-fragment (32x16 bf16): lane holds col N=lane%16, K=(lane/16)*16+0..15;
// 16 contiguous halfs when tile stored [N][K].
__device__ __forceinline__ v16bf load_b_frag(const unsigned short* p) {
    BF16Frag f;
    u16x8 lo = *(const u16x8*)(p);
    u16x8 hi = *(const u16x8*)(p + 8);
#pragma unroll
    for (int e = 0; e < 8; ++e) { f.u[e] = lo[e]; f.u[8 + e] = hi[e]; }
    return f.v;
}

// ---------------------------------------------------------------------------
// Kernel 1: xqkv = x @ wqkv  (fp32 in, bf16 WMMA, fused RoPE epilogue)
// ---------------------------------------------------------------------------
__global__ __launch_bounds__(256) void qkv_rope_gemm(
    const float* __restrict__ x,        // [MROWS, DMODEL]
    const float* __restrict__ wqkv,     // [DMODEL, QKVN]
    unsigned short* __restrict__ Qb,
    unsigned short* __restrict__ Kb,
    unsigned short* __restrict__ Vb)
{
    __shared__ __attribute__((aligned(16))) unsigned short lA[128 * 32];   // [m][k]
    __shared__ __attribute__((aligned(16))) unsigned short lB[128 * 32];   // [n][k]

    const int tid  = threadIdx.x;
    const int lane = tid & 31;
    const int wave = tid >> 5;
    const int wrow = wave >> 1;
    const int wcol = wave & 1;
    const int n0   = blockIdx.x * 128;
    const int m0   = blockIdx.y * 128;
    const int row16 = lane & 15;
    const int half  = lane >> 4;
    const int koff  = half * 8;

    v8f acc[2][4];
#pragma unroll
    for (int mt = 0; mt < 2; ++mt)
#pragma unroll
        for (int nt = 0; nt < 4; ++nt) acc[mt][nt] = (v8f)0.0f;

    for (int k0 = 0; k0 < DMODEL; k0 += 32) {
        __syncthreads();
        if (k0 + 32 < DMODEL) {   // prefetch next k-tile (weights dominate HBM BW)
            __builtin_prefetch(x + (size_t)(m0 + (tid >> 1)) * DMODEL + k0 + 32, 0, 1);
            __builtin_prefetch(wqkv + (size_t)(k0 + 32 + (tid & 31)) * QKVN + n0, 0, 1);
        }
        {   // stage A tile (fp32 -> bf16)
            int r  = tid >> 1;
            int ch = (tid & 1) * 16;
            const float* g = x + (size_t)(m0 + r) * DMODEL + k0 + ch;
#pragma unroll
            for (int i = 0; i < 16; ++i) lA[r * 32 + ch + i] = f2bf(g[i]);
        }
        {   // stage B tile transposed -> [n][k]
            int kk = tid & 31;
            int ng = tid >> 5;
            const float* g = wqkv + (size_t)(k0 + kk) * QKVN + n0 + ng * 16;
#pragma unroll
            for (int i = 0; i < 16; ++i) lB[(ng * 16 + i) * 32 + kk] = f2bf(g[i]);
        }
        __syncthreads();

        v16bf bfr[4];
#pragma unroll
        for (int nt = 0; nt < 4; ++nt)
            bfr[nt] = load_b_frag(&lB[(wcol * 64 + nt * 16 + row16) * 32 + half * 16]);
        v16bf a0 = load_a_frag(&lA[(wrow * 32 + 0  + row16) * 32], koff);
        v16bf a1 = load_a_frag(&lA[(wrow * 32 + 16 + row16) * 32], koff);
#pragma unroll
        for (int nt = 0; nt < 4; ++nt) {
            acc[0][nt] = wmma_bf16(a0, bfr[nt], acc[0][nt]);
            acc[1][nt] = wmma_bf16(a1, bfr[nt], acc[1][nt]);
        }
    }

    // Epilogue: RoPE on Q/K columns, scatter to head-major bf16 buffers.
#pragma unroll
    for (int mt = 0; mt < 2; ++mt) {
#pragma unroll
        for (int nt = 0; nt < 4; ++nt) {
#pragma unroll
            for (int i = 0; i < 8; ++i) {
                int gm = m0 + wrow * 32 + mt * 16 + i + 8 * half;
                int gn = n0 + wcol * 64 + nt * 16 + row16;
                int b  = gm >> 11;
                int s  = gm & (SEQ - 1);
                float v = acc[mt][nt][i];
                float other = __shfl_xor(v, 1, 32);
                float outv = v;
                if (gn < QDIM + KVDIM) {
                    int d2 = gn & (HEADDIM - 1) & ~1;
                    float ang = (float)s * exp2f(-(float)d2 * (LOG2_THETA / (float)HEADDIM));
                    float sn, cs;
                    __sincosf(ang, &sn, &cs);
                    if ((gn & 1) == 0) outv = v * cs - other * sn;
                    else               outv = other * sn + v * cs;
                }
                unsigned short ob = f2bf(outv);
                if (gn < QDIM) {
                    int h = gn >> 7, d = gn & 127;
                    Qb[((size_t)(b * NHEADS + h) * SEQ + s) * HEADDIM + d] = ob;
                } else if (gn < QDIM + KVDIM) {
                    int h = (gn - QDIM) >> 7, d = gn & 127;
                    Kb[((size_t)(b * NKVHEADS + h) * SEQ + s) * HEADDIM + d] = ob;
                } else {
                    int h = (gn - QDIM - KVDIM) >> 7, d = gn & 127;
                    Vb[((size_t)(b * NKVHEADS + h) * SEQ + s) * HEADDIM + d] = ob;
                }
            }
        }
    }
}

// ---------------------------------------------------------------------------
// Kernel 2: causal GQA flash-attention, KV tile = 64.
// grid = (B*NHEADS, SEQ/64), block = 128 threads (4 waves, 16 Q-rows each).
// K tile staged with GLOBAL_LOAD_ASYNC_TO_LDS_B128 (ASYNCcnt path).
// ---------------------------------------------------------------------------
__global__ __launch_bounds__(128) void flash_attn(
    const unsigned short* __restrict__ Qb,
    const unsigned short* __restrict__ Kb,
    const unsigned short* __restrict__ Vb,
    unsigned short* __restrict__ Ab,          // [MROWS, QDIM] bf16
    const int* __restrict__ is_causal_p)
{
    __shared__ __attribute__((aligned(16))) unsigned short lK[64 * 128];  // [kv][hd]
    __shared__ __attribute__((aligned(16))) unsigned short lV[128 * 64];  // [hd][kv]
    __shared__ __attribute__((aligned(16))) unsigned short lP[4][16 * 64];

    const int tid   = threadIdx.x;
    const int lane  = tid & 31;
    const int wave  = tid >> 5;
    const int row16 = lane & 15;
    const int half  = lane >> 4;
    const int koff  = half * 8;

    const int bh  = blockIdx.x;
    const int b   = bh >> 5;
    const int h   = bh & 31;
    const int kvh = h >> 2;
    const int q0  = blockIdx.y * 64;
    const int qbase = q0 + wave * 16;

    const unsigned short* Qg = Qb + ((size_t)bh * SEQ + qbase + row16) * HEADDIM;
    v16bf aq[4];
#pragma unroll
    for (int f = 0; f < 4; ++f) aq[f] = load_a_frag(Qg + f * 32, koff);

    v8f   o[8];
    float mprev[8], lsum[8];
#pragma unroll
    for (int f = 0; f < 8; ++f) o[f] = (v8f)0.0f;
#pragma unroll
    for (int i = 0; i < 8; ++i) { mprev[i] = -3.0e38f; lsum[i] = 0.0f; }

    const int causal = *is_causal_p;
    const int ntiles = causal ? (q0 / 64 + 1) : (SEQ / 64);
    const unsigned short* Kg0 = Kb + (size_t)(b * NKVHEADS + kvh) * SEQ * HEADDIM;
    const unsigned short* Vg0 = Vb + (size_t)(b * NKVHEADS + kvh) * SEQ * HEADDIM;

    for (int jt = 0; jt < ntiles; ++jt) {
        const int k0 = jt * 64;
        __syncthreads();
        {   // stage K via async global->LDS (128B stripe per thread)
            int r = tid >> 1;                 // 0..63 kv row
            int c = (tid & 1) * 64;           // hd half
            const unsigned short* gk = Kg0 + (size_t)(k0 + r) * HEADDIM + c;
            async_copy_128B((unsigned)(size_t)&lK[r * 128 + c],
                            (unsigned long long)(size_t)gk);
            // stage V transposed [hd][kv] (manual: async can't transpose)
            const unsigned short* gv = Vg0 + (size_t)(k0 + r) * HEADDIM + c;
#pragma unroll
            for (int i = 0; i < 64; ++i) lV[(c + i) * 64 + r] = gv[i];
        }
        wait_async0();
        __syncthreads();

        // scores: 16x64 = 4 x (16x16) tiles, contraction hd=128 -> 4 WMMAs each
        v8f sc[4];
#pragma unroll
        for (int nt = 0; nt < 4; ++nt) sc[nt] = (v8f)0.0f;
#pragma unroll
        for (int nt = 0; nt < 4; ++nt) {
#pragma unroll
            for (int f = 0; f < 4; ++f) {
                v16bf bk = load_b_frag(&lK[(nt * 16 + row16) * 128 + f * 32 + half * 16]);
                sc[nt] = wmma_bf16(aq[f], bk, sc[nt]);
            }
        }

        // online softmax (row = i + 8*half, col = k0 + nt*16 + row16)
#pragma unroll
        for (int i = 0; i < 8; ++i) {
            int qs = qbase + i + 8 * half;
            float mx = -3.0e38f;
#pragma unroll
            for (int nt = 0; nt < 4; ++nt) {
                int kv = k0 + nt * 16 + row16;
                float v = sc[nt][i] * INV_SQRT_HD;
                if (causal && kv > qs) v = -3.0e38f;
                sc[nt][i] = v;
                mx = fmaxf(mx, v);
            }
#pragma unroll
            for (int m = 1; m <= 8; m <<= 1)
                mx = fmaxf(mx, __shfl_xor(mx, m, 32));
            float mnew  = fmaxf(mprev[i], mx);
            float alpha = __expf(mprev[i] - mnew);
            float rs = 0.0f;
            int pr = (i + 8 * half) * 64;
#pragma unroll
            for (int nt = 0; nt < 4; ++nt) {
                float p = __expf(sc[nt][i] - mnew);
                rs += p;
                lP[wave][pr + nt * 16 + row16] = f2bf(p);
            }
#pragma unroll
            for (int m = 1; m <= 8; m <<= 1)
                rs += __shfl_xor(rs, m, 32);
            lsum[i]  = lsum[i] * alpha + rs;
            mprev[i] = mnew;
#pragma unroll
            for (int f = 0; f < 8; ++f) o[f][i] *= alpha;
        }
        __syncthreads();

        // O += P @ V  (contraction kv=64 -> 2 A-frags; 8 hd tiles of 16)
        v16bf ap0 = load_a_frag(&lP[wave][row16 * 64], koff);
        v16bf ap1 = load_a_frag(&lP[wave][row16 * 64 + 32], koff);
#pragma unroll
        for (int f = 0; f < 8; ++f) {
            v16bf bv0 = load_b_frag(&lV[(f * 16 + row16) * 64 + half * 16]);
            v16bf bv1 = load_b_frag(&lV[(f * 16 + row16) * 64 + 32 + half * 16]);
            o[f] = wmma_bf16(ap0, bv0, o[f]);
            o[f] = wmma_bf16(ap1, bv1, o[f]);
        }
    }

    // normalize and store bf16 attn output [b*SEQ+s][h*128+d]
#pragma unroll
    for (int i = 0; i < 8; ++i) {
        float linv = 1.0f / lsum[i];
        int s = qbase + i + 8 * half;
        size_t base = ((size_t)(b * SEQ + s)) * QDIM + h * HEADDIM;
#pragma unroll
        for (int f = 0; f < 8; ++f)
            Ab[base + f * 16 + row16] = f2bf(o[f][i] * linv);
    }
}

// ---------------------------------------------------------------------------
// Kernel 3: out = attn_out(bf16) @ wo(fp32->bf16), fp32 result.
// A tile staged via async global->LDS (already bf16).
// ---------------------------------------------------------------------------
__global__ __launch_bounds__(256) void out_gemm(
    const unsigned short* __restrict__ Ab,
    const float* __restrict__ wo,
    float* __restrict__ out)
{
    __shared__ __attribute__((aligned(16))) unsigned short lA[128 * 32];
    __shared__ __attribute__((aligned(16))) unsigned short lB[128 * 32];

    const int tid  = threadIdx.x;
    const int lane = tid & 31;
    const int wave = tid >> 5;
    const int wrow = wave >> 1;
    const int wcol = wave & 1;
    const int n0   = blockIdx.x * 128;
    const int m0   = blockIdx.y * 128;
    const int row16 = lane & 15;
    const int half  = lane >> 4;
    const int koff  = half * 8;

    v8f acc[2][4];
#pragma unroll
    for (int mt = 0; mt < 2; ++mt)
#pragma unroll
        for (int nt = 0; nt < 4; ++nt) acc[mt][nt] = (v8f)0.0f;

    for (int k0 = 0; k0 < QDIM; k0 += 32) {
        __syncthreads();
        if (k0 + 32 < QDIM)
            __builtin_prefetch(wo + (size_t)(k0 + 32 + (tid & 31)) * DMODEL + n0, 0, 1);
        {   // A already bf16 -> async copy 32B per thread
            int r  = tid >> 1;
            int ch = (tid & 1) * 16;
            const unsigned short* g = Ab + (size_t)(m0 + r) * QDIM + k0 + ch;
            async_copy_32B((unsigned)(size_t)&lA[r * 32 + ch],
                           (unsigned long long)(size_t)g);
        }
        {   // B fp32 -> bf16, transposed [n][k]
            int kk = tid & 31;
            int ng = tid >> 5;
            const float* g = wo + (size_t)(k0 + kk) * DMODEL + n0 + ng * 16;
#pragma unroll
            for (int i = 0; i < 16; ++i) lB[(ng * 16 + i) * 32 + kk] = f2bf(g[i]);
        }
        wait_async0();
        __syncthreads();

        v16bf bfr[4];
#pragma unroll
        for (int nt = 0; nt < 4; ++nt)
            bfr[nt] = load_b_frag(&lB[(wcol * 64 + nt * 16 + row16) * 32 + half * 16]);
        v16bf a0 = load_a_frag(&lA[(wrow * 32 + 0  + row16) * 32], koff);
        v16bf a1 = load_a_frag(&lA[(wrow * 32 + 16 + row16) * 32], koff);
#pragma unroll
        for (int nt = 0; nt < 4; ++nt) {
            acc[0][nt] = wmma_bf16(a0, bfr[nt], acc[0][nt]);
            acc[1][nt] = wmma_bf16(a1, bfr[nt], acc[1][nt]);
        }
    }

#pragma unroll
    for (int mt = 0; mt < 2; ++mt)
#pragma unroll
        for (int nt = 0; nt < 4; ++nt)
#pragma unroll
            for (int i = 0; i < 8; ++i) {
                int gm = m0 + wrow * 32 + mt * 16 + i + 8 * half;
                int gn = n0 + wcol * 64 + nt * 16 + row16;
                out[(size_t)gm * DMODEL + gn] = acc[mt][nt][i];
            }
}

// ---------------------------------------------------------------------------
// Host launcher
// ---------------------------------------------------------------------------
extern "C" void kernel_launch(void* const* d_in, const int* in_sizes, int n_in,
                              void* d_out, int out_size, void* d_ws, size_t ws_size,
                              hipStream_t stream) {
    const float* x     = (const float*)d_in[0];
    // d_in[1] = freqs_cis (recomputed on the fly in-kernel)
    const float* wqkv  = (const float*)d_in[2];
    const float* wo    = (const float*)d_in[3];
    const int* is_causal = (const int*)d_in[4];
    float* out = (float*)d_out;

    char* ws = (char*)d_ws;
    unsigned short* Qb = (unsigned short*)(ws);                       // 32 MB
    unsigned short* Kb = (unsigned short*)(ws + (size_t)33554432);    // 8 MB
    unsigned short* Vb = (unsigned short*)(ws + (size_t)41943040);    // 8 MB
    unsigned short* Ab = (unsigned short*)(ws + (size_t)50331648);    // 32 MB

    qkv_rope_gemm<<<dim3(QKVN / 128, MROWS / 128), 256, 0, stream>>>(
        x, wqkv, Qb, Kb, Vb);

    flash_attn<<<dim3(BATCH * NHEADS, SEQ / 64), 128, 0, stream>>>(
        Qb, Kb, Vb, Ab, is_causal);

    out_gemm<<<dim3(DMODEL / 128, MROWS / 128), 256, 0, stream>>>(
        Ab, wo, out);
}
